// LSTM_45037027066066
// MI455X (gfx1250) — matrix operand order repaired and checked by reading, not compiled, not run on hardware
//
#include <hip/hip_runtime.h>
#include <hip/hip_bf16.h>

// ---------------------------------------------------------------------------
// 6-layer stacked LSTM for MI455X (gfx1250, wave32, WMMA bf16 path)
// B=32, T=512, I=4096, H=64, 4H=256, L=6, OUT=4
// ---------------------------------------------------------------------------

typedef __attribute__((ext_vector_type(16))) __bf16 v16bf;
typedef __attribute__((ext_vector_type(8)))  float  v8f;

#define TB   512
#define NI   4096
#define NH   64
#define N4H  256
#define NB   32
#define NROWS (NB * TB)     // 16384 flattened (b,t) rows

__device__ __forceinline__ __bf16 f2bf(float f) { return (__bf16)f; }
__device__ __forceinline__ float sigmoidf(float x) { return 1.0f / (1.0f + __expf(-x)); }

// ---------------------------------------------------------------------------
// fp32 -> bf16 weight conversion (one pass; makes GEMM B-operand half-size)
// ---------------------------------------------------------------------------
__global__ void __launch_bounds__(256) cvt_bf16(const float* __restrict__ W,
                                                __bf16* __restrict__ Wb, int n) {
  int i = blockIdx.x * 256 + threadIdx.x;
  if (i < n) Wb[i] = f2bf(W[i]);
}

// ---------------------------------------------------------------------------
// Input-projection GEMM:  C[row, 0..255] = A[row, 0..K) @ Wb^T + bias
//   A:  [NROWS, K] fp32 row-major (lda == K)
//   Wb: [256, K]  bf16 row-major (pre-converted PyTorch weight)
// One workgroup (8 waves, 256 thr) computes a 64-row x 256-col slab, staging
// 64x32 A (fp32->bf16) and 256x32 B (bf16) chunks in LDS. Wave w computes
// M-tiles {2*(w>>2), 2*(w>>2)+1} x N-tiles [4*(w&3) .. +4): 8 WMMAs / chunk.
// ---------------------------------------------------------------------------
__global__ void __launch_bounds__(256) gemm_xp(const float* __restrict__ A,
                                               const __bf16* __restrict__ Wb,
                                               const float* __restrict__ bias,
                                               float* __restrict__ C, int K) {
  const int tid  = threadIdx.x;
  const int wave = tid >> 5;
  const int lane = tid & 31;
  const int half = lane >> 4;
  const int ln   = lane & 15;
  const int rowbase = blockIdx.x * 64;
  const int mt = wave >> 2;           // 0..1 -> rows [32*mt, 32*mt+32)
  const int nq = wave & 3;            // 0..3 -> cols [64*nq, 64*nq+64)

  __shared__ __bf16 Alds[64 * 34];    // 64x32 chunk, padded row stride 34
  __shared__ __bf16 Blds[256 * 34];   // 256 cols x 32 k, padded row stride 34

  v8f acc[2][4];
  #pragma unroll
  for (int mi = 0; mi < 2; ++mi)
    #pragma unroll
    for (int nt = 0; nt < 4; ++nt) acc[mi][nt] = (v8f){};

  const unsigned* __restrict__ Wbu = (const unsigned*)Wb;
  unsigned* Bldsu = (unsigned*)Blds;

  for (int k0 = 0; k0 < K; k0 += 32) {
    // stage A: 64x32 fp32 -> bf16 (2048 elements / 256 threads)
    {
      int idx = tid;
      #pragma unroll
      for (int r = 0; r < 8; ++r) {
        int m = idx >> 5, k = idx & 31;
        Alds[m * 34 + k] = f2bf(A[(size_t)(rowbase + m) * K + k0 + k]);
        idx += 256;
      }
    }
    // stage B: thread t copies col n=t's 32 bf16 (16 dwords), contiguous
    {
      const unsigned* src = &Wbu[((size_t)tid * K + k0) >> 1];
      unsigned* dst = &Bldsu[tid * 17];
      #pragma unroll
      for (int u = 0; u < 16; ++u) dst[u] = src[u];
    }
    __syncthreads();

    // A fragments (16x32 bf16 layout, ISA 7.12.2)
    v16bf af[2];
    #pragma unroll
    for (int mi = 0; mi < 2; ++mi) {
      int r = mt * 32 + mi * 16 + ln;
      #pragma unroll
      for (int j = 0; j < 8; ++j) {
        int kk = ((j >> 2) << 4) + 8 * half + 2 * (j & 3);
        af[mi][2 * j]     = Alds[r * 34 + kk];
        af[mi][2 * j + 1] = Alds[r * 34 + kk + 1];
      }
    }
    // B fragments (32x16 layout: lane = n + 16*half, VGPR j -> k = 16*(j>>2)+8*half+2*(j&3))
    v16bf bf[4];
    #pragma unroll
    for (int nt = 0; nt < 4; ++nt) {
      int n = nq * 64 + nt * 16 + ln;
      #pragma unroll
      for (int j = 0; j < 8; ++j) {
        int kk = ((j >> 2) << 4) + 8 * half + 2 * (j & 3);
        bf[nt][2 * j]     = Blds[n * 34 + kk];
        bf[nt][2 * j + 1] = Blds[n * 34 + kk + 1];
      }
    }

    #pragma unroll
    for (int mi = 0; mi < 2; ++mi)
      #pragma unroll
      for (int nt = 0; nt < 4; ++nt)
        acc[mi][nt] = __builtin_amdgcn_wmma_f32_16x16x32_bf16(
            false, af[mi], false, bf[nt], (short)0, acc[mi][nt], false, false);

    __syncthreads();
  }

  // epilogue: C/D f32 layout (VGPR v -> row v + 8*half, col = ln)
  #pragma unroll
  for (int mi = 0; mi < 2; ++mi) {
    #pragma unroll
    for (int nt = 0; nt < 4; ++nt) {
      int col = nq * 64 + nt * 16 + ln;
      float bv = bias[col];
      #pragma unroll
      for (int v = 0; v < 8; ++v) {
        int m = mt * 32 + mi * 16 + v + 8 * half;
        C[(size_t)(rowbase + m) * N4H + col] = acc[mi][nt][v] + bv;
      }
    }
  }
}

// ---------------------------------------------------------------------------
// Persistent per-layer LSTM recurrence.
// Grid = 2 workgroups x 128 threads (4 waves). WG g owns batches [16g,16g+16).
// Wave w owns gate w (columns [64w, 64w+64) of the 4H gate vector).
// Whh fragments preloaded into VGPRs; h (bf16) and c (fp32) live in LDS.
// Per step: g = xp[b,t,:] + h_prev @ Whh^T   via 2 chained bf16 WMMAs/tile.
// ---------------------------------------------------------------------------
__global__ void __launch_bounds__(128) lstm_rec(const float* __restrict__ xp,
                                                const float* __restrict__ Whh,
                                                float* __restrict__ hseq) {
  const int tid  = threadIdx.x;
  const int wave = tid >> 5;      // 0..3 == gate index (i, f, g, o)
  const int lane = tid & 31;
  const int half = lane >> 4;
  const int ln   = lane & 15;
  const int b0   = blockIdx.x * 16;

  __shared__ __bf16 hA[16 * 66];        // h_prev as bf16, padded stride 66
  __shared__ float  cbuf[16 * 64];      // cell state, fp32
  __shared__ float  gbuf[4][16 * 64];   // gate pre-activations

  // Preload Whh fragments: wave w, tile j covers cols 64w+16j .. +15, K=64.
  v16bf bfr[4][2];
  #pragma unroll
  for (int j = 0; j < 4; ++j) {
    int ng = wave * 64 + j * 16 + ln;
    #pragma unroll
    for (int kc = 0; kc < 2; ++kc) {
      #pragma unroll
      for (int jj = 0; jj < 8; ++jj) {
        int kk = kc * 32 + ((jj >> 2) << 4) + 8 * half + 2 * (jj & 3);
        const float* p = &Whh[(size_t)ng * NH + kk];
        bfr[j][kc][2 * jj]     = f2bf(p[0]);
        bfr[j][kc][2 * jj + 1] = f2bf(p[1]);
      }
    }
  }

  for (int e = tid; e < 16 * 64; e += 128) {
    cbuf[e] = 0.0f;
    hA[(e >> 6) * 66 + (e & 63)] = (__bf16)0.0f;
  }
  __syncthreads();

  for (int t = 0; t < TB; ++t) {
    // A fragments of h_prev (16x64 -> two 16x32 k-chunks)
    v16bf af0, af1;
    #pragma unroll
    for (int j = 0; j < 8; ++j) {
      int kk = ((j >> 2) << 4) + 8 * half + 2 * (j & 3);
      af0[2 * j]     = hA[ln * 66 + kk];
      af0[2 * j + 1] = hA[ln * 66 + kk + 1];
      af1[2 * j]     = hA[ln * 66 + 32 + kk];
      af1[2 * j + 1] = hA[ln * 66 + 32 + kk + 1];
    }

    #pragma unroll
    for (int j = 0; j < 4; ++j) {
      int colg = wave * 64 + j * 16 + ln;
      // seed accumulator with the precomputed input projection xp[b, t, colg]
      v8f acc;
      #pragma unroll
      for (int v = 0; v < 8; ++v) {
        int m = v + 8 * half;
        acc[v] = xp[((size_t)(b0 + m) * TB + t) * N4H + colg];
      }
      acc = __builtin_amdgcn_wmma_f32_16x16x32_bf16(false, af0, false, bfr[j][0],
                                                    (short)0, acc, false, false);
      acc = __builtin_amdgcn_wmma_f32_16x16x32_bf16(false, af1, false, bfr[j][1],
                                                    (short)0, acc, false, false);
      #pragma unroll
      for (int v = 0; v < 8; ++v) {
        int m = v + 8 * half;
        gbuf[wave][m * 64 + j * 16 + ln] = acc[v];
      }
    }
    __syncthreads();

    // gate nonlinearities + state update (1024 elems over 128 threads)
    for (int e = tid; e < 16 * 64; e += 128) {
      float gi = gbuf[0][e], gf = gbuf[1][e], gg = gbuf[2][e], go = gbuf[3][e];
      float c = sigmoidf(gf) * cbuf[e] + sigmoidf(gi) * tanhf(gg);
      cbuf[e] = c;
      float h = sigmoidf(go) * tanhf(c);
      int m = e >> 6, k = e & 63;
      hA[m * 66 + k] = f2bf(h);
      hseq[((size_t)(b0 + m) * TB + t) * NH + k] = h;
    }
    __syncthreads();
  }
}

// ---------------------------------------------------------------------------
// Output head: out[b, o] = h_last[b] . Wout[o] + bout[o]   (32x4 outputs)
// ---------------------------------------------------------------------------
__global__ void __launch_bounds__(128) out_proj(const float* __restrict__ hseq,
                                                const float* __restrict__ Wout,
                                                const float* __restrict__ bout,
                                                float* __restrict__ out) {
  int tid = threadIdx.x;
  int b = tid >> 2, o = tid & 3;
  const float* h = &hseq[((size_t)b * TB + (TB - 1)) * NH];
  float s = bout[o];
  #pragma unroll
  for (int k = 0; k < NH; ++k) s += h[k] * Wout[o * NH + k];
  out[b * 4 + o] = s;
}

// ---------------------------------------------------------------------------
extern "C" void kernel_launch(void* const* d_in, const int* in_sizes, int n_in,
                              void* d_out, int out_size, void* d_ws, size_t ws_size,
                              hipStream_t stream) {
  const float* x    = (const float*)d_in[0];   // [B,T,I]
  const float* Wih0 = (const float*)d_in[1];   // [4H,I]
  const float* Whh0 = (const float*)d_in[2];   // [4H,H]
  const float* b0   = (const float*)d_in[3];   // [4H]
  const float* WihR = (const float*)d_in[4];   // [L-1,4H,H]
  const float* WhhR = (const float*)d_in[5];   // [L-1,4H,H]
  const float* bR   = (const float*)d_in[6];   // [L-1,4H]
  const float* Wout = (const float*)d_in[7];   // [OUT,H]
  const float* bout = (const float*)d_in[8];   // [OUT]

  float*  xp   = (float*)d_ws;                       // [NROWS, 256]  16 MB
  float*  hseq = xp + (size_t)NROWS * N4H;           // [NROWS, 64]    4 MB
  __bf16* Wb   = (__bf16*)(hseq + (size_t)NROWS * NH); // [256, K<=4096] 2 MB

  // Layer 0: convert Wih0 to bf16, big input projection (K = 4096), recurrence.
  cvt_bf16<<<(N4H * NI + 255) / 256, 256, 0, stream>>>(Wih0, Wb, N4H * NI);
  gemm_xp<<<NROWS / 64, 256, 0, stream>>>(x, Wb, b0, xp, NI);
  lstm_rec<<<2, 128, 0, stream>>>(xp, Whh0, hseq);

  // Layers 1..5: K = 64 projection of previous layer's h-sequence, recurrence.
  for (int l = 0; l < 5; ++l) {
    cvt_bf16<<<(N4H * NH + 255) / 256, 256, 0, stream>>>(
        WihR + (size_t)l * N4H * NH, Wb, N4H * NH);
    gemm_xp<<<NROWS / 64, 256, 0, stream>>>(hseq, Wb,
                                            bR + (size_t)l * N4H, xp, NH);
    lstm_rec<<<2, 128, 0, stream>>>(xp, WhhR + (size_t)l * N4H * NH, hseq);
  }

  out_proj<<<1, 128, 0, stream>>>(hseq, Wout, bout, (float*)d_out);
}
